// RNN_24575802868508
// MI455X (gfx1250) — matrix operand order repaired and checked by reading, not compiled, not run on hardware
//
#include <hip/hip_runtime.h>

// ---------------------------------------------------------------------------
// LRN (2-layer light recurrent network), T=512 B=64 I=H=1024.
// GEMM: bf16 V_WMMA_F32_16X16X32_BF16, tiles fed by the Tensor Data Mover
//       (tensor_load_to_lds) into double-buffered LDS, waits on TENSORcnt.
// Scan: streaming elementwise recurrence, one thread per (b, h).
// ---------------------------------------------------------------------------

typedef __attribute__((ext_vector_type(16))) __bf16 v16bf;
typedef __attribute__((ext_vector_type(8)))  float  v8f;
typedef unsigned int __attribute__((ext_vector_type(4))) uint4v;
typedef int          __attribute__((ext_vector_type(8))) int8v;
typedef int          __attribute__((ext_vector_type(4))) int4v;

#define T_DIM 512
#define B_DIM 64
#define I_DIM 1024
#define H_DIM 1024

#define BM 128
#define BN 128
#define BK 32
// TDM pads 4 DWORDs after every 16 DWORDs -> effective LDS row stride
// 32 + 8 bf16 = 40 (20*n mod 64 hits distinct banks for lanes 0..15).
#define LDS_STRIDE 40
#define TILE_USHORTS (BM * LDS_STRIDE)   // 5120 ushorts = 10240 B per tile

__device__ __forceinline__ unsigned short f2bf(float f) {
    unsigned u = __float_as_uint(f);
    unsigned r = (u + 0x7FFFu + ((u >> 16) & 1u)) >> 16;   // round-to-nearest-even
    return (unsigned short)r;
}

// --------------------------- fp32 -> bf16 ----------------------------------
__global__ void f32_to_bf16_kernel(const float* __restrict__ src,
                                   unsigned short* __restrict__ dst, size_t n) {
    size_t i = (size_t)blockIdx.x * blockDim.x + threadIdx.x;
    size_t stride = (size_t)gridDim.x * blockDim.x;
    for (; i < n; i += stride) dst[i] = f2bf(src[i]);
}

// ---------------- fp32 W[K,N] -> bf16 Wt[N,K] (convert + transpose) --------
// One-off 6 MB transform per layer; lets the B tile be a plain row-major
// (N x K) 2D tile: TDM-loadable and ds_load_b128-fragmentable like A.
__global__ void w_to_bf16_tr_kernel(const float* __restrict__ W,
                                    unsigned short* __restrict__ Wt,
                                    int K, int N) {
    size_t total = (size_t)K * N;
    size_t i = (size_t)blockIdx.x * blockDim.x + threadIdx.x;
    size_t stride = (size_t)gridDim.x * blockDim.x;
    for (; i < total; i += stride) {
        int k = (int)(i / N);
        int n = (int)(i - (size_t)k * N);
        Wt[(size_t)n * K + k] = f2bf(W[i]);
    }
}

// --------------------- TDM 2D tile load (global -> LDS) --------------------
// Builds a D# per CDNA5 ISA and issues
// TENSOR_LOAD_TO_LDS. 2D tensor: groups 2/3 are NULL-equivalent (zeros).
// Padding: pad_interval code 3 (every 16 DWORDs) + pad_amount code 3
// (4 DWORDs) => row stride 32 bf16 + 8 bf16 pad = LDS_STRIDE.
// This toolchain exposes the 6-arg builtin:
//   (uint32x4 g0, int32x8 g1, int32x4 g2, int32x4 g3, int32x8 g4, i32 cpol)
__device__ __forceinline__ void tdm_load_tile_2d(
    const unsigned short* gbase,  // tile start in global memory
    unsigned lds_off,             // LDS byte offset of destination buffer
    unsigned tensor_d0,           // row length (elements)
    unsigned tensor_d1,           // number of rows
    unsigned tile_d0,             // tile row length (elements)
    unsigned tile_d1)             // tile rows
{
    unsigned long long ga = (unsigned long long)(size_t)gbase;
    uint4v g0;
    g0.x = 1u;                                                // count=1, user mode
    g0.y = lds_off;                                           // lds_addr [63:32]
    g0.z = (unsigned)(ga & 0xFFFFFFFFu);                      // global_addr lo
    g0.w = (unsigned)((ga >> 32) & 0x01FFFFFFu) | 0x80000000u;// addr[56:32] | type=2

    int8v g1;
    g1[0] = (int)((1u << 16)        // data_size = 2 bytes
                | (1u << 20)        // pad_enable
                | (3u << 22)        // pad_interval: every 16 DWORDs
                | (3u << 25));      // pad_amount: 4 DWORDs
    g1[1] = (int)((tensor_d0 & 0xFFFFu) << 16);               // dim0[15:0] @ [63:48]
    g1[2] = (int)((tensor_d0 >> 16) | ((tensor_d1 & 0xFFFFu) << 16));
    g1[3] = (int)((tensor_d1 >> 16) | (tile_d0 << 16));       // tile_dim0 @ [127:112]
    g1[4] = (int)(tile_d1 & 0xFFFFu);                         // tile_dim1; tile_dim2=0
    g1[5] = (int)tensor_d0;                                   // dim0_stride (dense rows)
    g1[6] = 0;
    g1[7] = 0;

    int4v z4;
    z4[0] = 0; z4[1] = 0; z4[2] = 0; z4[3] = 0;
    int8v z8;
    z8[0] = 0; z8[1] = 0; z8[2] = 0; z8[3] = 0;
    z8[4] = 0; z8[5] = 0; z8[6] = 0; z8[7] = 0;
    __builtin_amdgcn_tensor_load_to_lds(g0, g1, z4, z4, z8, 0);
}

// --------------------------- WMMA bf16 GEMM --------------------------------
// C[M,N](f32) = A[M,K](bf16 row-major) * Bt[N,K](bf16 row-major)^T + bias[N]
// 256 threads = 8 waves; block tile 128x128; waves 2(M) x 4(N); each wave
// owns 64x32 = 4x2 grid of 16x16 f32 accumulators. Double-buffered LDS fed
// by the TDM (wave 0 issues, TENSORcnt + barrier synchronize).
__global__ __launch_bounds__(256) void gemm_bf16_wmma_kernel(
    const unsigned short* __restrict__ A,
    const unsigned short* __restrict__ Bt,
    const float* __restrict__ bias,
    float* __restrict__ C,
    int M, int N, int K)
{
    __shared__ unsigned short lA[2][TILE_USHORTS];   // [row 0..127][k 0..31] pad 40
    __shared__ unsigned short lB[2][TILE_USHORTS];   // [n   0..127][k 0..31] pad 40

    const int tid   = threadIdx.x;
    const int wave  = tid >> 5;
    const int lane  = tid & 31;
    const int waveM = wave >> 2;     // 0..1  (64 rows each)
    const int waveN = wave & 3;      // 0..3  (32 cols each)
    const int m0    = blockIdx.y * BM;
    const int n0    = blockIdx.x * BN;

    const unsigned ldsA0 = (unsigned)(unsigned long long)(size_t)(void*)&lA[0][0];
    const unsigned ldsA1 = (unsigned)(unsigned long long)(size_t)(void*)&lA[1][0];
    const unsigned ldsB0 = (unsigned)(unsigned long long)(size_t)(void*)&lB[0][0];
    const unsigned ldsB1 = (unsigned)(unsigned long long)(size_t)(void*)&lB[1][0];

    v8f zero = {};
    v8f acc[4][2];
    #pragma unroll
    for (int mi = 0; mi < 4; ++mi)
        #pragma unroll
        for (int ni = 0; ni < 2; ++ni) acc[mi][ni] = zero;

    union Frag { v16bf v; uint4 u[2]; };

    const int nk = K / BK;

    // prologue: DMA tile 0 into buffer 0
    if (wave == 0) {
        tdm_load_tile_2d(A  + (size_t)m0 * K, ldsA0, K, M, BK, BM);
        tdm_load_tile_2d(Bt + (size_t)n0 * K, ldsB0, K, N, BK, BN);
    }
    __builtin_amdgcn_s_wait_tensorcnt(0);
    __syncthreads();

    for (int kt = 0; kt < nk; ++kt) {
        const int cur = kt & 1;
        // DMA next tile into the other buffer while we compute this one.
        // (Other buffer was last read in iter kt-1; the barrier at the end of
        //  kt-1 guarantees all waves are done with it.)
        if (kt + 1 < nk && wave == 0) {
            const int k1 = (kt + 1) * BK;
            tdm_load_tile_2d(A  + (size_t)m0 * K + k1, cur ? ldsA0 : ldsA1, K, M, BK, BM);
            tdm_load_tile_2d(Bt + (size_t)n0 * K + k1, cur ? ldsB0 : ldsB1, K, N, BK, BN);
        }

        // ---- fragments per ISA 16-bit layouts (all contiguous b128 reads) ----
        const int r = lane & 15;
        Frag fa[4], fb[2];
        const int kbA = (lane < 16) ? 0 : 8;   // A: lanes0-15 K{0..7,16..23}; 16-31 K{8..15,24..31}
        #pragma unroll
        for (int mi = 0; mi < 4; ++mi) {
            int row = waveM * 64 + mi * 16 + r;
            fa[mi].u[0] = *(const uint4*)(&lA[cur][row * LDS_STRIDE + kbA]);
            fa[mi].u[1] = *(const uint4*)(&lA[cur][row * LDS_STRIDE + kbA + 16]);
        }
        const int kbB = (lane < 16) ? 0 : 8;   // B: lanes0-15 K{0..7,16..23} of col n; 16-31 K{8..15,24..31}
        #pragma unroll
        for (int ni = 0; ni < 2; ++ni) {
            int col = waveN * 32 + ni * 16 + r;
            fb[ni].u[0] = *(const uint4*)(&lB[cur][col * LDS_STRIDE + kbB]);
            fb[ni].u[1] = *(const uint4*)(&lB[cur][col * LDS_STRIDE + kbB + 16]);
        }
        #pragma unroll
        for (int mi = 0; mi < 4; ++mi)
            #pragma unroll
            for (int ni = 0; ni < 2; ++ni)
                acc[mi][ni] = __builtin_amdgcn_wmma_f32_16x16x32_bf16(
                    /*neg_a=*/false, fa[mi].v, /*neg_b=*/false, fb[ni].v,
                    /*c_mod=*/(short)0, acc[mi][ni],
                    /*reuse_a=*/false, /*reuse_b=*/false);

        // next tile landed (wave0's TENSORcnt; others' count is already 0),
        // and everyone is done reading the current buffer.
        __builtin_amdgcn_s_wait_tensorcnt(0);
        __syncthreads();
    }

    // ---- epilogue: C layout (VGPR g: lanes0-15 -> M=g, lanes16-31 -> M=8+g) ----
    const int r     = lane & 15;
    const int mhalf = (lane >> 4) * 8;
    #pragma unroll
    for (int mi = 0; mi < 4; ++mi) {
        #pragma unroll
        for (int ni = 0; ni < 2; ++ni) {
            int col  = n0 + waveN * 32 + ni * 16 + r;
            float bv = bias[col];
            #pragma unroll
            for (int g = 0; g < 8; ++g) {
                int row = m0 + waveM * 64 + mi * 16 + mhalf + g;
                C[(size_t)row * N + col] = acc[mi][ni][g] + bv;
            }
        }
    }
}

// ----------------------------- recurrence scan -----------------------------
// gates: [T, B, 3H] f32 (p|q|r). One thread per (b, h); loops over T.
__global__ __launch_bounds__(256) void lrn_scan_kernel(
    const float* __restrict__ gates,
    const float* __restrict__ h0,          // [B, H]
    const int*   __restrict__ L,           // [B]
    unsigned short* __restrict__ out_bf,   // [T, B, H] bf16 or nullptr
    float* __restrict__ out_f,             // [T, B, H] f32  or nullptr
    float* __restrict__ h_last,            // [B, H]
    int T, int B, int H)
{
    int idx = blockIdx.x * blockDim.x + threadIdx.x;     // b*H + hh
    int b   = idx / H;
    int hh  = idx - b * H;
    float h  = h0[idx];
    int   Lb = L[b];
    const size_t strideT = (size_t)B * 3 * H;
    const float* gp = gates + (size_t)b * 3 * H + hh;

    for (int t = 0; t < T; ++t) {
        float p  = gp[0];
        float q  = gp[H];
        float rr = gp[2 * H];
        float ig = 1.f / (1.f + __expf(-(p + h)));
        float fg = 1.f / (1.f + __expf(-(q - h)));
        float nh = tanhf(ig * rr + fg * h);
        if (t < Lb) h = nh;                 // mask: hold state past sequence end
        size_t o = (size_t)t * B * H + idx;
        if (out_bf) out_bf[o] = f2bf(h);
        if (out_f)  out_f[o]  = h;
        gp += strideT;
    }
    h_last[idx] = h;
}

// ------------------------------- launcher ----------------------------------
extern "C" void kernel_launch(void* const* d_in, const int* in_sizes, int n_in,
                              void* d_out, int out_size, void* d_ws, size_t ws_size,
                              hipStream_t stream) {
    (void)in_sizes; (void)n_in; (void)out_size; (void)ws_size;
    const int T = T_DIM, B = B_DIM, I = I_DIM, H = H_DIM;

    const float* x  = (const float*)d_in[0];   // [T, B, I]
    const float* h0 = (const float*)d_in[1];   // [2, B, H]
    const float* W1 = (const float*)d_in[2];   // [I, 3H]
    const float* b1 = (const float*)d_in[3];   // [3H]
    const float* W2 = (const float*)d_in[4];   // [H, 3H]
    const float* b2 = (const float*)d_in[5];   // [3H]
    const int*   L  = (const int*)d_in[6];     // [B]
    float* out = (float*)d_out;

    // workspace layout
    char* ws = (char*)d_ws;
    float*          gates = (float*)ws;          ws += (size_t)T * B * 3 * H * sizeof(float);
    unsigned short* xbf   = (unsigned short*)ws; ws += (size_t)T * B * I * sizeof(unsigned short);
    unsigned short* o1bf  = (unsigned short*)ws; ws += (size_t)T * B * H * sizeof(unsigned short);
    unsigned short* wtbf  = (unsigned short*)ws; // Wt: [3H, K] bf16, reused for W1/W2

    // d_out layout: out2 [T,B,H] ++ h_last1 [B,H] ++ h_last2 [B,H]
    float* out2   = out;
    float* hlast1 = out + (size_t)T * B * H;
    float* hlast2 = hlast1 + (size_t)B * H;

    const int M = T * B, N = 3 * H, K = I;     // 32768 x 3072 x 1024 (both layers)
    dim3 gemmGrid(N / BN, M / BM);             // 24 x 256 blocks

    // Layer 1
    f32_to_bf16_kernel<<<2048, 256, 0, stream>>>(x, xbf, (size_t)T * B * I);
    w_to_bf16_tr_kernel<<<2048, 256, 0, stream>>>(W1, wtbf, K, N);
    gemm_bf16_wmma_kernel<<<gemmGrid, 256, 0, stream>>>(xbf, wtbf, b1, gates, M, N, K);
    lrn_scan_kernel<<<(B * H) / 256, 256, 0, stream>>>(gates, h0, L,
                                                       o1bf, nullptr, hlast1, T, B, H);
    // Layer 2 (A operand = layer-1 output, already bf16)
    w_to_bf16_tr_kernel<<<2048, 256, 0, stream>>>(W2, wtbf, K, N);
    gemm_bf16_wmma_kernel<<<gemmGrid, 256, 0, stream>>>(o1bf, wtbf, b2, gates, M, N, K);
    lrn_scan_kernel<<<(B * H) / 256, 256, 0, stream>>>(gates, h0 + (size_t)B * H, L,
                                                       nullptr, out2, hlast2, T, B, H);
}
